// MultiHeadAttention_55791625175356
// MI455X (gfx1250) — matrix operand order repaired and checked by reading, not compiled, not run on hardware
//
#include <hip/hip_runtime.h>
#include <hip/hip_bf16.h>
#include <math.h>

// ---------------------------------------------------------------------------
// MI455X (gfx1250) causal multi-head attention forward, f16 WMMA pipeline.
// Round 2: GEMM wave tile enlarged 32x32 -> 64x64 (4x4 WMMA tiles) to double
// register-level arithmetic intensity (16 WMMAs per 8 fragment loads).
// ---------------------------------------------------------------------------

typedef _Float16 half_t;
typedef __attribute__((ext_vector_type(16))) _Float16 v16h;
typedef __attribute__((ext_vector_type(8)))  _Float16 v8h;
typedef __attribute__((ext_vector_type(8)))  float    v8f;

#define NHEAD  16
#define DMODEL 1024
#define BATCH  4
#define SEQ    2048
#define HDIM   64

// ---- WMMA fragment loaders (wave32, 16x16x32 f16) -------------------------
// A-matrix 16x32 (ISA 7.12.2): lane l holds row (l&15); halves are
// K = hi*8 + {0..7} and K = hi*8 + 16 + {0..7}, hi = l>>4.
__device__ __forceinline__ v16h load_fragA(const half_t* __restrict__ base, int ldk) {
  const int lane = threadIdx.x & 31;
  const int r = lane & 15, hi = lane >> 4;
  const half_t* p = base + (size_t)r * ldk + hi * 8;
  v8h x0 = *(const v8h*)(p);
  v8h x1 = *(const v8h*)(p + 16);
  return __builtin_shufflevector(x0, x1, 0,1,2,3,4,5,6,7,8,9,10,11,12,13,14,15);
}

// B-matrix 32x16: lane l holds column (l&15); halves are
// K = hi*8 + {0..7} and K = hi*8 + 16 + {0..7}, hi = l>>4. Source stored [N][K].
__device__ __forceinline__ v16h load_fragB(const half_t* __restrict__ base, int ldk) {
  const int lane = threadIdx.x & 31;
  const int r = lane & 15, hi = lane >> 4;
  const half_t* p = base + (size_t)r * ldk + hi * 8;
  v8h x0 = *(const v8h*)(p);
  v8h x1 = *(const v8h*)(p + 16);
  return __builtin_shufflevector(x0, x1, 0,1,2,3,4,5,6,7,8,9,10,11,12,13,14,15);
}

__device__ __forceinline__ v8f wmma32(v16h a, v16h b, v8f c) {
  return __builtin_amdgcn_wmma_f32_16x16x32_f16(false, a, false, b, (short)0, c,
                                                false, false);
}

__device__ __forceinline__ float redmax16(float v) {
  v = fmaxf(v, __shfl_xor(v, 1, 32));
  v = fmaxf(v, __shfl_xor(v, 2, 32));
  v = fmaxf(v, __shfl_xor(v, 4, 32));
  v = fmaxf(v, __shfl_xor(v, 8, 32));
  return v;
}
__device__ __forceinline__ float redsum16(float v) {
  v += __shfl_xor(v, 1, 32);
  v += __shfl_xor(v, 2, 32);
  v += __shfl_xor(v, 4, 32);
  v += __shfl_xor(v, 8, 32);
  return v;
}

// ---- prep kernels ---------------------------------------------------------
__global__ void cvt_f32_f16(const float* __restrict__ src, half_t* __restrict__ dst, int n) {
  int i = blockIdx.x * blockDim.x + threadIdx.x;
  if (i < n) dst[i] = (half_t)src[i];
}

// W[k][n] (f32) -> Wt[n][k] (f16), so WMMA B fragments load contiguously.
__global__ void transpose_cvt(const float* __restrict__ W, half_t* __restrict__ Wt,
                              int K, int N) {
  int i = blockIdx.x * blockDim.x + threadIdx.x;
  if (i < K * N) {
    int k = i / N, n = i - k * N;
    Wt[(size_t)n * K + k] = (half_t)W[i];
  }
}

// ---- shared WMMA GEMM mainloop: 64x64 per wave (4x4 tiles), K step 32 -----
// 16 WMMAs per 8 fragment loads -> 2.0 WMMA per fragment (vs 1.0 at 32x32).
__device__ __forceinline__ void gemm_mainloop(const half_t* __restrict__ A,
                                              const half_t* __restrict__ Bm,
                                              int K, v8f acc[4][4]) {
  for (int k = 0; k < K; k += 32) {
    v16h a[4], b[4];
#pragma unroll
    for (int i = 0; i < 4; ++i) a[i] = load_fragA(A + (size_t)(16 * i) * K + k, K);
#pragma unroll
    for (int j = 0; j < 4; ++j) b[j] = load_fragB(Bm + (size_t)(16 * j) * K + k, K);
#pragma unroll
    for (int i = 0; i < 4; ++i)
#pragma unroll
      for (int j = 0; j < 4; ++j)
        acc[i][j] = wmma32(a[i], b[j], acc[i][j]);
  }
}

// ---- stage 1: qkv = x @ W_qkv + b_qkv, scattered to attention layouts -----
// Qh, Kh : [B*H][SEQ][HDIM] f16 (row-major)
// Vt     : [B*H][HDIM][SEQ] f16 (transposed, so PV B-fragments are contiguous)
__global__ void __launch_bounds__(256)
qkv_gemm(const half_t* __restrict__ xh, const half_t* __restrict__ Wt,
         const float* __restrict__ bias, half_t* __restrict__ Qh,
         half_t* __restrict__ Kh, half_t* __restrict__ Vt) {
  const int wid = threadIdx.x >> 5, lane = threadIdx.x & 31;
  const int r = lane & 15, hi = lane >> 4;
  const int wm = wid >> 2, wn = wid & 3;              // 2 x 4 wave grid
  const int m0 = blockIdx.y * 128 + wm * 64;
  const int n0 = blockIdx.x * 256 + wn * 64;

  v8f acc[4][4] = {};
  gemm_mainloop(xh + (size_t)m0 * DMODEL, Wt + (size_t)n0 * DMODEL, DMODEL, acc);

#pragma unroll
  for (int tm = 0; tm < 4; ++tm) {
#pragma unroll
    for (int tn = 0; tn < 4; ++tn) {
      const int gn = n0 + tn * 16 + r;
      const float bv = bias[gn];
      const int sec = gn >> 10;           // 0=Q, 1=K, 2=V
      const int d = gn & (DMODEL - 1);
      const int h = d >> 6, f = d & (HDIM - 1);
#pragma unroll
      for (int v = 0; v < 8; ++v) {
        const int gm = m0 + tm * 16 + v + 8 * hi;
        const int b = gm >> 11, c = gm & (SEQ - 1);
        const half_t val = (half_t)(acc[tm][tn][v] + bv);
        const size_t bh = (size_t)(b * NHEAD + h);
        if (sec == 0)      Qh[(bh * SEQ + c) * HDIM + f] = val;
        else if (sec == 1) Kh[(bh * SEQ + c) * HDIM + f] = val;
        else               Vt[(bh * HDIM + f) * SEQ + c] = val;
      }
    }
  }
}

// ---- stage 2: causal flash attention, one 16-row q-tile per wave ----------
__global__ void __launch_bounds__(128)
flash_attn(const half_t* __restrict__ Qh, const half_t* __restrict__ Kh,
           const half_t* __restrict__ Vt, half_t* __restrict__ attn) {
  __shared__ __align__(16) half_t Pbuf[4][16 * 32];   // per-wave P bounce buffer

  const int bh = blockIdx.y;                 // b*H + h
  const int b = bh >> 4, h = bh & 15;
  const int wid = threadIdx.x >> 5, lane = threadIdx.x & 31;
  const int r = lane & 15, hi = lane >> 4;
  const int q0 = blockIdx.x * 64 + wid * 16; // this wave's q-tile base

  const half_t* Qb = Qh + ((size_t)bh * SEQ + q0) * HDIM;
  const half_t* Kb = Kh + (size_t)bh * SEQ * HDIM;
  const half_t* Vb = Vt + (size_t)bh * HDIM * SEQ;
  half_t* P = Pbuf[wid];

  // Q fragments are loop-invariant (hd=64 -> two k=32 chunks)
  const v16h qf0 = load_fragA(Qb, HDIM);
  const v16h qf1 = load_fragA(Qb + 32, HDIM);

  v8f acc0 = {}, acc1 = {}, acc2 = {}, acc3 = {};     // O: 16 x 64 fp32
  float mrow[8], lrow[8], corr[8];
#pragma unroll
  for (int v = 0; v < 8; ++v) { mrow[v] = -__builtin_inff(); lrow[v] = 0.f; }

  const float scale = 0.125f;               // 1/sqrt(64)
  const int kend = q0 + 16;                 // causal: keys < q0+16

  for (int kb = 0; kb < kend; kb += 32) {
    // S = Q K^T for 32 keys: two 16-key column tiles, two feature chunks each
    v8f s0 = {}, s1 = {};
    {
      v16h k00 = load_fragB(Kb + (size_t)kb * HDIM, HDIM);
      v16h k01 = load_fragB(Kb + (size_t)kb * HDIM + 32, HDIM);
      s0 = wmma32(qf0, k00, s0);
      s0 = wmma32(qf1, k01, s0);
      v16h k10 = load_fragB(Kb + (size_t)(kb + 16) * HDIM, HDIM);
      v16h k11 = load_fragB(Kb + (size_t)(kb + 16) * HDIM + 32, HDIM);
      s1 = wmma32(qf0, k10, s1);
      s1 = wmma32(qf1, k11, s1);
    }

    // online softmax: each 16-lane half holds rows v+8*hi, one column per lane
    const int kk0 = kb + r, kk1 = kb + 16 + r;
#pragma unroll
    for (int v = 0; v < 8; ++v) {
      const int qrow = q0 + v + 8 * hi;
      float a0 = (kk0 <= qrow) ? s0[v] * scale : -__builtin_inff();
      float a1 = (kk1 <= qrow) ? s1[v] * scale : -__builtin_inff();
      float tmax = redmax16(fmaxf(a0, a1));
      float newm = fmaxf(mrow[v], tmax);
      float p0 = __expf(a0 - newm);
      float p1 = __expf(a1 - newm);
      float rs = redsum16(p0 + p1);
      float cf = __expf(mrow[v] - newm);
      lrow[v] = lrow[v] * cf + rs;
      mrow[v] = newm;
      corr[v] = cf;
      const int row = v + 8 * hi;
      P[row * 32 + r]      = (half_t)p0;
      P[row * 32 + 16 + r] = (half_t)p1;
    }

    // rescale running O by per-row correction
#pragma unroll
    for (int v = 0; v < 8; ++v) {
      acc0[v] *= corr[v]; acc1[v] *= corr[v];
      acc2[v] *= corr[v]; acc3[v] *= corr[v];
    }

    // Same-wave LDS store->load (DS is in-order per wave); fence the compiler.
    __builtin_amdgcn_wave_barrier();
    const v16h pf = load_fragA(P, 32);
    __builtin_amdgcn_wave_barrier();

    // O += P V : V^T rows are features, keys contiguous -> B fragments
    v16h vf0 = load_fragB(Vb + (size_t)0  * SEQ + kb, SEQ);
    v16h vf1 = load_fragB(Vb + (size_t)16 * SEQ + kb, SEQ);
    v16h vf2 = load_fragB(Vb + (size_t)32 * SEQ + kb, SEQ);
    v16h vf3 = load_fragB(Vb + (size_t)48 * SEQ + kb, SEQ);
    acc0 = wmma32(pf, vf0, acc0);
    acc1 = wmma32(pf, vf1, acc1);
    acc2 = wmma32(pf, vf2, acc2);
    acc3 = wmma32(pf, vf3, acc3);
  }

  // normalize and store attn (f16, [B][C][D] with head concat) for O-proj
#pragma unroll
  for (int v = 0; v < 8; ++v) {
    const float inv = 1.0f / lrow[v];
    const size_t o = ((size_t)b * SEQ + q0 + v + 8 * hi) * DMODEL + h * HDIM + r;
    attn[o]      = (half_t)(acc0[v] * inv);
    attn[o + 16] = (half_t)(acc1[v] * inv);
    attn[o + 32] = (half_t)(acc2[v] * inv);
    attn[o + 48] = (half_t)(acc3[v] * inv);
  }
}

// ---- stage 3: out = attn @ W_o + b_o (fp32 output) ------------------------
__global__ void __launch_bounds__(256)
oproj_gemm(const half_t* __restrict__ Ah, const half_t* __restrict__ Wt,
           const float* __restrict__ bias, float* __restrict__ out) {
  const int wid = threadIdx.x >> 5, lane = threadIdx.x & 31;
  const int r = lane & 15, hi = lane >> 4;
  const int wm = wid >> 2, wn = wid & 3;
  const int m0 = blockIdx.y * 128 + wm * 64;
  const int n0 = blockIdx.x * 256 + wn * 64;

  v8f acc[4][4] = {};
  gemm_mainloop(Ah + (size_t)m0 * DMODEL, Wt + (size_t)n0 * DMODEL, DMODEL, acc);

#pragma unroll
  for (int tm = 0; tm < 4; ++tm) {
#pragma unroll
    for (int tn = 0; tn < 4; ++tn) {
      const int gn = n0 + tn * 16 + r;
      const float bv = bias[gn];
#pragma unroll
      for (int v = 0; v < 8; ++v) {
        const int gm = m0 + tm * 16 + v + 8 * hi;
        out[(size_t)gm * DMODEL + gn] = acc[tm][tn][v] + bv;
      }
    }
  }
}

// ---------------------------------------------------------------------------
extern "C" void kernel_launch(void* const* d_in, const int* in_sizes, int n_in,
                              void* d_out, int out_size, void* d_ws, size_t ws_size,
                              hipStream_t stream) {
  const float* x     = (const float*)d_in[0];   // [B,C,D]
  const float* W_qkv = (const float*)d_in[1];   // [D,3D]
  const float* b_qkv = (const float*)d_in[2];   // [3D]
  const float* W_o   = (const float*)d_in[3];   // [D,D]
  const float* b_o   = (const float*)d_in[4];   // [D]
  float* out = (float*)d_out;

  const size_t MB = 1024u * 1024u;
  char* ws = (char*)d_ws;
  half_t* xh    = (half_t*)(ws + 0 * MB);   // 16 MiB : x as f16
  half_t* Wtqkv = (half_t*)(ws + 16 * MB);  //  6 MiB : W_qkv^T f16 [3D][D]
  half_t* Wto   = (half_t*)(ws + 22 * MB);  //  2 MiB : W_o^T  f16 [D][D]
  half_t* Qh    = (half_t*)(ws + 24 * MB);  // 16 MiB : [B*H][C][64]
  half_t* Kh    = (half_t*)(ws + 40 * MB);  // 16 MiB : [B*H][C][64]
  half_t* Vt    = (half_t*)(ws + 56 * MB);  // 16 MiB : [B*H][64][C]
  half_t* attn  = (half_t*)(ws + 72 * MB);  // 16 MiB : [B][C][D]   (total 88 MiB)

  // prep
  {
    int n = BATCH * SEQ * DMODEL;
    cvt_f32_f16<<<(n + 255) / 256, 256, 0, stream>>>(x, xh, n);
    int nq = DMODEL * 3 * DMODEL;
    transpose_cvt<<<(nq + 255) / 256, 256, 0, stream>>>(W_qkv, Wtqkv, DMODEL, 3 * DMODEL);
    int no = DMODEL * DMODEL;
    transpose_cvt<<<(no + 255) / 256, 256, 0, stream>>>(W_o, Wto, DMODEL, DMODEL);
  }

  // QKV projection: M=8192, N=3072, K=1024 (block tile 128x256, wave 64x64)
  {
    dim3 grid(3 * DMODEL / 256, BATCH * SEQ / 128);
    qkv_gemm<<<grid, 256, 0, stream>>>(xh, Wtqkv, b_qkv, Qh, Kh, Vt);
  }

  // causal flash attention: 64 q rows per block, 4 waves of 16 rows
  {
    dim3 grid(SEQ / 64, BATCH * NHEAD);
    flash_attn<<<grid, 128, 0, stream>>>(Qh, Kh, Vt, attn);
  }

  // output projection: M=8192, N=1024, K=1024 (block tile 128x256)
  {
    dim3 grid(DMODEL / 256, BATCH * SEQ / 128);
    oproj_gemm<<<grid, 256, 0, stream>>>(attn, Wto, b_o, out);
  }
}
